// ESpaceLoss_89249420410957
// MI455X (gfx1250) — matrix operand order repaired and checked by reading, not compiled, not run on hardware
//
#include <hip/hip_runtime.h>
#include <hip/hip_bf16.h>
#include <math.h>

typedef __attribute__((ext_vector_type(16))) _Float16 v16h;
typedef __attribute__((ext_vector_type(8)))  float    v8f;

#define T_TOTAL 48000
#define T_MID   16000   // middle third [16000, 32000)
#define NCH     64
#define NPROBE  128     // fwd + bwd channels
#define NOUT    64
#define NB      4

// workspace layout:
//   hbuf: [which(2)][b(4)][t(16000)][c(128)]  f16   = 32,768,000 bytes
//   acc : float[8]  {lin0,log0,lin1,log1,lin2,log2, align_sum, raw_sum}
//   Wh  : f16 copy of W (64x128) at HBYTES + 256
#define HBYTES ((size_t)2 * NB * T_MID * NPROBE * sizeof(_Float16))

// Native CDNA5 V_TANH_F32 if the toolchain exposes it; branch-free fallback.
__device__ __forceinline__ float fast_tanh(float x) {
#if __has_builtin(__builtin_amdgcn_tanhf)
  return __builtin_amdgcn_tanhf(x);
#elif __has_builtin(__builtin_amdgcn_tanh_f32)
  return __builtin_amdgcn_tanh_f32(x);
#else
  // tanh(x) = 1 - 2/(exp(2x)+1); __expf -> v_exp_f32 (no exec-mask branches)
  float e = __expf(2.0f * x);
  return 1.0f - 2.0f / (e + 1.0f);
#endif
}

// ---------------------------------------------------------------------------
// 1) IIR probe with 4-step lookahead.
//    State s_t = M s_{t-1} + b0 x_t e1, M = [[-a1,-a2],[1,0]].
//    Per group of 4 samples only y_{t+2}, y_{t+3} (2 dependent FMAs each) are on the
//    serial critical path; x-convolution terms and the other 2 outputs are
//    independent work -> ~4x shorter dependency chain on the only
//    latency-bound kernel. 1024 chains (which,b,dir,ch), f64 as reference.
// ---------------------------------------------------------------------------
__global__ void iir_probe_kernel(const float* __restrict__ pred,
                                 const float* __restrict__ target,
                                 _Float16* __restrict__ hbuf) {
  int tid = blockIdx.x * blockDim.x + threadIdx.x;
  if (tid >= 1024) return;
  int ch    = tid & 63;
  int dir   = (tid >> 6) & 1;   // 0 = fwd, 1 = bwd (time-reversed)
  int b     = (tid >> 7) & 3;
  int which = (tid >> 9) & 1;   // 0 = pred, 1 = target

  // filter coefficients (match reference, f64)
  double lf = log(10.0) + (log(100.0) - log(10.0)) * (double)ch / 63.0;
  double f  = exp(lf);
  double r  = 0.9999 + (0.99999 - 0.9999) * (double)ch / 63.0;
  double th = 2.0 * 3.14159265358979323846 * f / 24000.0;
  double a1 = -2.0 * r * cos(th);
  double a2 = r * r;
  double b0 = (1.0 - r) * 0.5;

  // powers of the companion matrix M = [[-a1,-a2],[1,0]]
  double m00 = -a1, m01 = -a2;
  // M^1
  double p00 = m00, p01 = m01, p10 = 1.0, p11 = 0.0;
  // M^2 = M * M^1
  double q00 = m00 * p00 + m01 * p10, q01 = m00 * p01 + m01 * p11;
  double q10 = p00,                   q11 = p01;
  // M^3 = M * M^2
  double r00 = m00 * q00 + m01 * q10, r01 = m00 * q01 + m01 * q11;
  double r10 = q00,                   r11 = q01;
  // M^4 = M * M^3
  double s00 = m00 * r00 + m01 * r10, s01 = m00 * r01 + m01 * r11;

  // g_i = (M^i)_{00} : input-convolution taps
  double g1 = p00, g2 = q00, g3 = r00;
  // c_j = (M^{j+1})_{00,01} : state taps for y_{t+j}
  double c00 = p00, c01 = p01;
  double c10 = q00, c11 = q01;
  double c20 = r00, c21 = r01;
  double c30 = s00, c31 = s01;

  const float* x = (which ? target : pred) + (size_t)b * T_TOTAL;
  int c = dir * NCH + ch;
  _Float16* hout = hbuf + ((size_t)(which * NB + b) * T_MID) * NPROBE + c;

  double y1 = 0.0, y2 = 0.0;  // y_{t-1}, y_{t-2}
  for (int k = 0; k < T_TOTAL; k += 4) {
    int i0 = dir ? (T_TOTAL - 1 - k) : k;
    int step = dir ? -1 : 1;
    double x0 = b0 * (double)x[i0];
    double x1v = b0 * (double)x[i0 + step];
    double x2v = b0 * (double)x[i0 + 2 * step];
    double x3v = b0 * (double)x[i0 + 3 * step];
    // input-convolution terms (off the critical path)
    double X0 = x0;
    double X1 = x1v + g1 * x0;
    double X2 = x2v + g1 * x1v + g2 * x0;
    double X3 = x3v + g1 * x2v + g2 * x1v + g3 * x0;
    // outputs: y_{k+j} = c_j0*y1 + c_j1*y2 + X_j
    double y0n = c00 * y1 + c01 * y2 + X0;
    double y1n = c10 * y1 + c11 * y2 + X1;
    double y2n = c20 * y1 + c21 * y2 + X2;
    double y3n = c30 * y1 + c31 * y2 + X3;  // this and y2n feed next group
    // store middle third
    unsigned tm;
    tm = (unsigned)(i0 - T_MID);
    if (tm < (unsigned)T_MID) hout[(size_t)tm * NPROBE] = (_Float16)(float)y0n;
    tm = (unsigned)(i0 + step - T_MID);
    if (tm < (unsigned)T_MID) hout[(size_t)tm * NPROBE] = (_Float16)(float)y1n;
    tm = (unsigned)(i0 + 2 * step - T_MID);
    if (tm < (unsigned)T_MID) hout[(size_t)tm * NPROBE] = (_Float16)(float)y2n;
    tm = (unsigned)(i0 + 3 * step - T_MID);
    if (tm < (unsigned)T_MID) hout[(size_t)tm * NPROBE] = (_Float16)(float)y3n;
    y2 = y2n;
    y1 = y3n;
  }
}

// ---------------------------------------------------------------------------
// 1b) one-shot f32 -> f16 conversion of W (removes cvt chains from WMMA loop)
// ---------------------------------------------------------------------------
__global__ void wprep_kernel(const float* __restrict__ W,
                             _Float16* __restrict__ Wh) {
  int i = blockIdx.x * blockDim.x + threadIdx.x;
  if (i < NOUT * NPROBE) Wh[i] = (_Float16)W[i];
}

// ---------------------------------------------------------------------------
// 2) WMMA embedding: emb = tanh(W(64x128) @ h(128 x t)), accumulate
//    |emb_pred - emb_target| over the middle third. One wave per 16x16 tile.
// ---------------------------------------------------------------------------
__global__ void embed_align_kernel(const _Float16* __restrict__ Wh,
                                   const _Float16* __restrict__ hbuf,
                                   float* __restrict__ acc) {
  int wave = (int)((blockIdx.x * blockDim.x + threadIdx.x) >> 5);
  int lane = threadIdx.x & 31;
  if (wave >= NB * 4 * (T_MID / 16)) return;   // wave-uniform guard

  int tt = wave % (T_MID / 16);
  int ot = (wave / (T_MID / 16)) & 3;
  int b  = wave / (4 * (T_MID / 16));
  int o0 = ot * 16;
  int t0 = tt * 16;
  int group = lane >> 4;    // 0: lanes 0-15, 1: lanes 16-31
  int mn    = lane & 15;    // row for A, column (t) for B/C/D

  const _Float16* hp = hbuf + (size_t)(0 * NB + b) * T_MID * NPROBE;
  const _Float16* ht = hbuf + (size_t)(1 * NB + b) * T_MID * NPROBE;

  __builtin_prefetch(Wh + (o0 + mn) * NPROBE, 0, 3);  // global_prefetch_b8

  v8f accp = {};
  v8f acct = {};

  for (int kk = 0; kk < 4; ++kk) {     // K = 128 in 4 steps of 32
    int c0 = kk * 32;
    v16h a, bp, bt;
#pragma unroll
    for (int v = 0; v < 8; ++v) {
#pragma unroll
      for (int hh = 0; hh < 2; ++hh) {
        // A (16x32 f16) layout per ISA: lanes0-15 V0..3 -> K0..7, V4..7 -> K16..23
        //                               lanes16-31 shift K by +8
        int ka = ((v < 4) ? 0 : 16) + group * 8 + (v & 3) * 2 + hh;
        // B (32x16 f16) layout: same K order as A
        int kb = ((v < 4) ? 0 : 16) + group * 8 + (v & 3) * 2 + hh;
        a[v * 2 + hh]  = Wh[(o0 + mn) * NPROBE + c0 + ka];
        bp[v * 2 + hh] = hp[(size_t)(t0 + mn) * NPROBE + c0 + kb];
        bt[v * 2 + hh] = ht[(size_t)(t0 + mn) * NPROBE + c0 + kb];
      }
    }
    accp = __builtin_amdgcn_wmma_f32_16x16x32_f16(
        false, a, false, bp, (short)0, accp, false, false);
    acct = __builtin_amdgcn_wmma_f32_16x16x32_f16(
        false, a, false, bt, (short)0, acct, false, false);
  }

  float s = 0.f;
#pragma unroll
  for (int r2 = 0; r2 < 8; ++r2)
    s += fabsf(fast_tanh(accp[r2]) - fast_tanh(acct[r2]));
  // wave32 reduction
  for (int off = 16; off > 0; off >>= 1) s += __shfl_xor(s, off, 32);
  if (lane == 0) atomicAdd(&acc[6], s);
}

// ---------------------------------------------------------------------------
// 3) STFT loss: one block per (frame, batch). pred frame -> real part,
//    target frame -> imag part, one complex radix-2 FFT in LDS, then unpack
//    both rfft spectra and fuse |mag| and |log mag| L1 into atomics.
// ---------------------------------------------------------------------------
template <int NFFT, int HOP, int ACCIDX>
__global__ void stft_loss_kernel(const float* __restrict__ pred,
                                 const float* __restrict__ target,
                                 float* __restrict__ acc) {
  constexpr int LOG2N = (NFFT == 512) ? 9 : (NFFT == 1024) ? 10 : 11;
  __shared__ float sre[NFFT];
  __shared__ float sim[NFFT];
  __shared__ float red[256];

  int frame = blockIdx.x;
  int b     = blockIdx.y;
  const float* xp = pred   + (size_t)b * T_TOTAL + (size_t)frame * HOP;
  const float* xt = target + (size_t)b * T_TOTAL + (size_t)frame * HOP;

  // bit-reversed load
  for (int i = threadIdx.x; i < NFFT; i += blockDim.x) {
    unsigned j = __brev((unsigned)i) >> (32 - LOG2N);
    sre[i] = xp[j];
    sim[i] = xt[j];
  }
  __syncthreads();

  // iterative radix-2 DIT
  for (int s = 1; s <= LOG2N; ++s) {
    int half = 1 << (s - 1);
    int mlen = 1 << s;
    for (int idx = threadIdx.x; idx < NFFT / 2; idx += blockDim.x) {
      int grp  = idx >> (s - 1);
      int j    = idx & (half - 1);
      int i0   = grp * mlen + j;
      int i1   = i0 + half;
      float ang = -6.28318530717958647692f * (float)j / (float)mlen;
      float wi, wr;
      __sincosf(ang, &wi, &wr);
      float ur = sre[i0], ui = sim[i0];
      float vr = sre[i1], vi = sim[i1];
      float tr = wr * vr - wi * vi;
      float ti = wr * vi + wi * vr;
      sre[i0] = ur + tr; sim[i0] = ui + ti;
      sre[i1] = ur - tr; sim[i1] = ui - ti;
    }
    __syncthreads();
  }

  // unpack the two real spectra and accumulate the losses
  float lin = 0.f, lg = 0.f;
  for (int k = threadIdx.x; k <= NFFT / 2; k += blockDim.x) {
    int nk = (NFFT - k) & (NFFT - 1);
    float zr = sre[k],  zi = sim[k];
    float z2r = sre[nk], z2i = sim[nk];
    float pre = 0.5f * (zr + z2r), pim = 0.5f * (zi - z2i);
    float tre = 0.5f * (zi + z2i), tim = 0.5f * (z2r - zr);
    float mp = sqrtf(pre * pre + pim * pim) + 1e-7f;
    float mt = sqrtf(tre * tre + tim * tim) + 1e-7f;
    lin += fabsf(mp - mt);
    lg  += fabsf(__logf(mp) - __logf(mt));
  }

  red[threadIdx.x] = lin;
  __syncthreads();
  for (int off = 128; off > 0; off >>= 1) {
    if ((int)threadIdx.x < off) red[threadIdx.x] += red[threadIdx.x + off];
    __syncthreads();
  }
  if (threadIdx.x == 0) atomicAdd(&acc[ACCIDX], red[0]);
  __syncthreads();

  red[threadIdx.x] = lg;
  __syncthreads();
  for (int off = 128; off > 0; off >>= 1) {
    if ((int)threadIdx.x < off) red[threadIdx.x] += red[threadIdx.x + off];
    __syncthreads();
  }
  if (threadIdx.x == 0) atomicAdd(&acc[ACCIDX + 1], red[0]);
}

// ---------------------------------------------------------------------------
// 4) Raw L1 over the middle third
// ---------------------------------------------------------------------------
__global__ void raw_loss_kernel(const float* __restrict__ pred,
                                const float* __restrict__ target,
                                float* __restrict__ acc) {
  __shared__ float red[256];
  int i = blockIdx.x * blockDim.x + threadIdx.x;
  int stride = gridDim.x * blockDim.x;
  float s = 0.f;
  for (int e = i; e < NB * T_MID; e += stride) {
    int b = e / T_MID;
    int t = T_MID + (e % T_MID);
    s += fabsf(pred[(size_t)b * T_TOTAL + t] - target[(size_t)b * T_TOTAL + t]);
  }
  red[threadIdx.x] = s;
  __syncthreads();
  for (int off = 128; off > 0; off >>= 1) {
    if ((int)threadIdx.x < off) red[threadIdx.x] += red[threadIdx.x + off];
    __syncthreads();
  }
  if (threadIdx.x == 0) atomicAdd(&acc[7], red[0]);
}

// ---------------------------------------------------------------------------
// 5) Finalize: combine the 8 accumulators into (total, l_mss, l_align)
// ---------------------------------------------------------------------------
__global__ void finalize_kernel(const float* __restrict__ acc,
                                float* __restrict__ out) {
  // frame counts: (48000 - n)/hop + 1 ; bins = n/2 + 1 ; x batch 4
  const float cnt0 = (float)(NB * 372 * 257);
  const float cnt1 = (float)(NB * 184 * 513);
  const float cnt2 = (float)(NB * 90 * 1025);
  float l_mss = (acc[0] + acc[1]) / cnt0 +
                (acc[2] + acc[3]) / cnt1 +
                (acc[4] + acc[5]) / cnt2;
  float l_align = acc[6] / (float)((size_t)NB * NOUT * T_MID);
  float l_raw   = acc[7] / (float)(NB * T_MID) * 0.1f;
  float total = l_mss + l_align * 10.0f + l_raw;
  out[0] = total;
  out[1] = l_mss;
  out[2] = l_align;
}

// ---------------------------------------------------------------------------
extern "C" void kernel_launch(void* const* d_in, const int* in_sizes, int n_in,
                              void* d_out, int out_size, void* d_ws,
                              size_t ws_size, hipStream_t stream) {
  (void)in_sizes; (void)n_in; (void)out_size; (void)ws_size;
  const float* pred   = (const float*)d_in[0];
  const float* target = (const float*)d_in[1];
  const float* W      = (const float*)d_in[2];

  _Float16* hbuf = (_Float16*)d_ws;
  float*    acc  = (float*)((char*)d_ws + HBYTES);
  _Float16* Wh   = (_Float16*)((char*)d_ws + HBYTES + 256);

  hipMemsetAsync(acc, 0, 8 * sizeof(float), stream);

  // W -> f16 once
  wprep_kernel<<<(NOUT * NPROBE + 255) / 256, 256, 0, stream>>>(W, Wh);

  // 1024 f64 IIR chains, lookahead-4 (critical path ~2 FMA / 4 samples)
  iir_probe_kernel<<<4, 256, 0, stream>>>(pred, target, hbuf);

  // 16000 16x16 WMMA tiles, 8 waves per block
  embed_align_kernel<<<2000, 256, 0, stream>>>(Wh, hbuf, acc);

  // multi-scale STFT losses (pred packed as Re, target as Im)
  stft_loss_kernel<512, 128, 0><<<dim3(372, NB), 256, 0, stream>>>(pred, target, acc);
  stft_loss_kernel<1024, 256, 2><<<dim3(184, NB), 256, 0, stream>>>(pred, target, acc);
  stft_loss_kernel<2048, 512, 4><<<dim3(90, NB), 256, 0, stream>>>(pred, target, acc);

  // raw L1
  raw_loss_kernel<<<64, 256, 0, stream>>>(pred, target, acc);

  // combine
  finalize_kernel<<<1, 1, 0, stream>>>(acc, (float*)d_out);
}
